// ImuEncoder_89455578841860
// MI455X (gfx1250) — matrix operand order repaired and checked by reading, not compiled, not run on hardware
//
#include <hip/hip_runtime.h>
#include <hip/hip_bf16.h>
#include <math.h>

// ---------------------------------------------------------------------------
// IMU encoder for MI455X (gfx1250): f16 WMMA pipeline
//   k_cvt : f32->f16 weight conversion (W2, W_ih, W_hh)
//   k1    : feats (6ch + dt) -> gelu(Linear 7->256)              [scalar]
//   k2    : gelu(x1 @ W2^T + b2) -> xg = x2 @ W_ih^T + b_ih      [WMMA]
//   k3    : GRU scan; gate-aligned WMMA strips, async xg staging [WMMA+async]
//   k4    : masked softmax attention pool + LayerNorm + Wp       [scalar]
// ---------------------------------------------------------------------------

// Flip to 0 if the gfx1250 MC rejects the async-to-LDS mnemonics.
#define USE_ASYNC 1

typedef _Float16 half_t;
typedef __attribute__((ext_vector_type(16))) _Float16 v16h;
typedef __attribute__((ext_vector_type(8)))  _Float16 v8h;
typedef __attribute__((ext_vector_type(8)))  float    v8f;

#define B_  128
#define T_  2048
#define H_  256
#define K_  3
#define D_  256
#define G3  (3 * H_)   // 768

// ----------------------------- device helpers ------------------------------

static __device__ __forceinline__ float gelu_exact(float x) {
  return 0.5f * x * (1.0f + erff(x * 0.70710678118654752440f));
}
static __device__ __forceinline__ float sigmoid_(float x) {
  return 1.0f / (1.0f + expf(-x));
}

// A-operand fragment, 16x32 f16, from row-major [.., ld] storage.
// ISA 7.12.2: lanes 0-15 hold M=0..15 with K pairs {0..7,16..23};
// lanes 16-31 hold M=0..15 with K pairs {8..15,24..31}.
static __device__ __forceinline__ v16h load_a_frag(const half_t* p, int ld, int lane) {
  const int m = lane & 15, hw = (lane >> 4) & 1;
  const half_t* r = p + (size_t)m * ld + hw * 8;
  union { v16h v; v8h h[2]; } u;
  u.h[0] = *(const v8h*)(r);        // K = hw*8 + 0..7
  u.h[1] = *(const v8h*)(r + 16);   // K = 16 + hw*8 + 0..7
  return u.v;
}

// B-operand fragment, 32x16 f16 (K x N) taken from row-major W[N][K]
// (i.e. B = W^T): lane holds column n = lane%16; halves carry
// K = (lane/16)*8 + 0..7 and 16 + (lane/16)*8 + 0..7 (same as A).
static __device__ __forceinline__ v16h load_b_frag(const half_t* p, int ld, int lane) {
  const int n = lane & 15, hw = (lane >> 4) & 1;
  const half_t* r = p + (size_t)n * ld + hw * 8;
  union { v16h v; v8h h[2]; } u;
  u.h[0] = *(const v8h*)(r);
  u.h[1] = *(const v8h*)(r + 16);
  return u.v;
}

static __device__ __forceinline__ v8f wmma16(v16h a, v16h b, v8f c) {
  return __builtin_amdgcn_wmma_f32_16x16x32_f16(
      /*neg_a=*/false, a, /*neg_b=*/false, b,
      /*c_mod=*/(short)0, c, /*reuse_a=*/false, /*reuse_b=*/false);
}

// One b128 (16B) global -> LDS transfer per active lane, tracked by ASYNCcnt.
static __device__ __forceinline__ void async_copy_b128(half_t* lds_dst,
                                                       const half_t* gsrc) {
#if USE_ASYNC
  const unsigned loff = (unsigned)(unsigned long long)lds_dst;  // LDS byte offset
  asm volatile("global_load_async_to_lds_b128 %0, %1, off"
               :: "v"(loff), "v"((unsigned long long)gsrc)
               : "memory");
#else
  *(v8h*)lds_dst       = *(const v8h*)gsrc;
  *(v8h*)(lds_dst + 8) = *(const v8h*)(gsrc + 8);
#endif
}
static __device__ __forceinline__ void async_wait0() {
#if USE_ASYNC
  asm volatile("s_wait_asynccnt 0" ::: "memory");
#endif
}

static __device__ __forceinline__ float block_sum256(float v, float* red) {
  const int tid = threadIdx.x;
  red[tid] = v; __syncthreads();
  for (int s = 128; s > 0; s >>= 1) {
    if (tid < s) red[tid] += red[tid + s];
    __syncthreads();
  }
  float r = red[0]; __syncthreads();
  return r;
}
static __device__ __forceinline__ float block_max256(float v, float* red) {
  const int tid = threadIdx.x;
  red[tid] = v; __syncthreads();
  for (int s = 128; s > 0; s >>= 1) {
    if (tid < s) red[tid] = fmaxf(red[tid], red[tid + s]);
    __syncthreads();
  }
  float r = red[0]; __syncthreads();
  return r;
}

// ------------------------------- kernels -----------------------------------

__global__ __launch_bounds__(256) void k_cvt(const float* __restrict__ src,
                                             half_t* __restrict__ dst, int n) {
  int i = blockIdx.x * 256 + threadIdx.x;
  if (i < n) dst[i] = (half_t)src[i];
}

// One block per flattened row (b*T + t); thread = output channel h.
__global__ __launch_bounds__(256) void k1_inproj1(const float* __restrict__ samples,
                                                  const float* __restrict__ W1,
                                                  const float* __restrict__ b1,
                                                  half_t* __restrict__ x1) {
  const long row = blockIdx.x;            // b*T + t
  const int  h   = threadIdx.x;
  const int  t   = (int)(row & (T_ - 1));
  const float* s = samples + row * 7;
  float f[7];
#pragma unroll
  for (int c = 0; c < 6; ++c) f[c] = s[c];
  f[6] = (t > 0) ? (s[6] - s[-1]) : 0.0f;   // s[-1] == samples[b][t-1][6]
  const float* w = W1 + h * 7;
  float a = b1[h];
#pragma unroll
  for (int c = 0; c < 7; ++c) a += w[c] * f[c];
  x1[row * H_ + h] = (half_t)gelu_exact(a);
}

// 16-row tile per block: x2 = gelu(x1 @ W2^T + b2); xg = x2 @ W_ih^T + b_ih.
__global__ __launch_bounds__(256) void k2_x2_xg(const half_t* __restrict__ x1,
                                                const half_t* __restrict__ W2h,
                                                const float*  __restrict__ b2,
                                                const half_t* __restrict__ Wihh,
                                                const float*  __restrict__ bih,
                                                half_t* __restrict__ xg) {
  __shared__ __align__(16) half_t x1s[16 * H_];
  __shared__ __align__(16) half_t x2s[16 * H_];
  const int tid  = threadIdx.x;
  const int lane = tid & 31, wave = tid >> 5;
  const long row0 = (long)blockIdx.x * 16;

  {  // stage the 16x256 f16 A-tile: 512 v8h chunks, 2 per thread
    const v8h* s = (const v8h*)(x1 + row0 * H_);
    v8h* d = (v8h*)x1s;
    d[tid]       = s[tid];
    d[tid + 256] = s[tid + 256];
  }
  __syncthreads();

  const int n0 = lane & 15, hw = lane >> 4;

  // layer 2: A fragments once, 16 N-tiles over 8 waves
  {
    v16h afr[8];
#pragma unroll
    for (int kt = 0; kt < 8; ++kt) afr[kt] = load_a_frag(x1s + kt * 32, H_, lane);
#pragma unroll
    for (int i = 0; i < 2; ++i) {
      const int nt = wave * 2 + i;
      v8f acc = {};
#pragma unroll
      for (int kt = 0; kt < 8; ++kt)
        acc = wmma16(afr[kt],
                     load_b_frag(W2h + (size_t)(nt * 16) * H_ + kt * 32, H_, lane),
                     acc);
      const int n = nt * 16 + n0;
      const float bias = b2[n];
#pragma unroll
      for (int r = 0; r < 8; ++r)
        x2s[(r + hw * 8) * H_ + n] = (half_t)gelu_exact(acc[r] + bias);
    }
  }
  __syncthreads();

  // input gates: 48 N-tiles over 8 waves
  {
    v16h afr[8];
#pragma unroll
    for (int kt = 0; kt < 8; ++kt) afr[kt] = load_a_frag(x2s + kt * 32, H_, lane);
#pragma unroll
    for (int i = 0; i < 6; ++i) {
      const int nt = wave * 6 + i;
      v8f acc = {};
#pragma unroll
      for (int kt = 0; kt < 8; ++kt)
        acc = wmma16(afr[kt],
                     load_b_frag(Wihh + (size_t)(nt * 16) * H_ + kt * 32, H_, lane),
                     acc);
      const int n = nt * 16 + n0;
      const float bias = bih[n];
#pragma unroll
      for (int r = 0; r < 8; ++r)
        xg[(size_t)(row0 + r + hw * 8) * G3 + n] = (half_t)(acc[r] + bias);
    }
  }
}

// GRU scan: one block per 16 batch rows. Wave w owns column strip
// [32w, 32w+32) of ALL THREE gates (r-tiles {2w,2w+1}, z {16+2w,..},
// n {32+2w,..}) so gates come straight out of the wave's accumulators.
// h ping-pongs in LDS; the xg gate slice is async-staged during the WMMAs.
__global__ __launch_bounds__(256) void k3_gru(const half_t* __restrict__ xg,
                                              const half_t* __restrict__ Whh,
                                              const float*  __restrict__ bhh_g,
                                              half_t* __restrict__ Hseq) {
  __shared__ __align__(16) half_t hsb[2][16 * H_];   // 16 KB ping-pong h (f16)
  __shared__ __align__(16) half_t xgs[16 * G3];      // 24 KB gate staging
  __shared__ float bhh[G3];                          // 3 KB
  const int tid  = threadIdx.x;
  const int lane = tid & 31, wave = tid >> 5;
  const int b0   = blockIdx.x * 16;

  for (int i = tid; i < 2 * 16 * H_; i += 256) (&hsb[0][0])[i] = (half_t)0.0f;
  for (int i = tid; i < G3; i += 256) bhh[i] = bhh_g[i];
  __syncthreads();

  const int n0 = lane & 15, hw = lane >> 4;
  const int colw = wave * 32;  // this wave's 32-column gate strip

  for (int t = 0; t < T_; ++t) {
    // 1) async-stage this wave's xg slice: 16 rows x 3 gates x 32 cols
    //    = 192 b128 chunks, 6 per lane. Overlaps the WMMA stage below.
#pragma unroll
    for (int c = 0; c < 6; ++c) {
      const int i = c * 32 + lane;
      const int m = i / 12, p = i % 12;
      const int col = (p >> 2) * H_ + colw + (p & 3) * 8;
      const size_t grow = ((size_t)(b0 + m) * T_ + t) * G3;
      async_copy_b128(&xgs[m * G3 + col], xg + grow + col);
    }

    // 2) A fragments from the current h buffer (held in registers)
    const half_t* hs = hsb[t & 1];
    v16h afr[8];
#pragma unroll
    for (int kt = 0; kt < 8; ++kt) afr[kt] = load_a_frag(hs + kt * 32, H_, lane);

    // 3) gate GEMM strips: gh = h @ W_hh^T for r/z/n column pairs
    v8f accr[2], accz[2], accn[2];
#pragma unroll
    for (int i = 0; i < 2; ++i) {
      const half_t* wr = Whh + (size_t)((2 * wave + i) * 16) * H_;
      const half_t* wz = wr + (size_t)H_ * H_;        // +16 tiles
      const half_t* wn = wr + (size_t)(2 * H_) * H_;  // +32 tiles
      accr[i] = (v8f){}; accz[i] = (v8f){}; accn[i] = (v8f){};
#pragma unroll
      for (int kt = 0; kt < 8; ++kt) {
        if (kt < 7) {
          __builtin_prefetch(wr + (kt + 1) * 32, 0, 0);
          __builtin_prefetch(wz + (kt + 1) * 32, 0, 0);
          __builtin_prefetch(wn + (kt + 1) * 32, 0, 0);
        }
        accr[i] = wmma16(afr[kt], load_b_frag(wr + kt * 32, H_, lane), accr[i]);
        accz[i] = wmma16(afr[kt], load_b_frag(wz + kt * 32, H_, lane), accz[i]);
        accn[i] = wmma16(afr[kt], load_b_frag(wn + kt * 32, H_, lane), accn[i]);
      }
    }

    // 4) gates from own accumulators (wave-local; no gh staging needed)
    async_wait0();  // this wave's xgs slice is resident
    half_t* hnext = hsb[(t + 1) & 1];
#pragma unroll
    for (int i = 0; i < 2; ++i) {
      const int col = colw + i * 16 + n0;
      const float br = bhh[col], bz = bhh[H_ + col], bn = bhh[2 * H_ + col];
#pragma unroll
      for (int r = 0; r < 8; ++r) {
        const int m = r + hw * 8;
        const float xr = (float)xgs[m * G3 + col];
        const float xz = (float)xgs[m * G3 + H_ + col];
        const float xn = (float)xgs[m * G3 + 2 * H_ + col];
        const float rg = sigmoid_(xr + accr[i][r] + br);
        const float zg = sigmoid_(xz + accz[i][r] + bz);
        const float cg = tanhf(xn + rg * (accn[i][r] + bn));
        const float hnew = (1.0f - zg) * cg + zg * (float)hs[m * H_ + col];
        hnext[m * H_ + col] = (half_t)hnew;
        Hseq[((size_t)(b0 + m) * T_ + t) * H_ + col] = (half_t)hnew;
      }
    }
    __syncthreads();  // publish hnext for next step's A fragments
  }
}

// Attention pool + LayerNorm + output projection; one block per batch.
__global__ __launch_bounds__(256) void k4_attn(const half_t* __restrict__ Hseq,
                                               const int*    __restrict__ lengths,
                                               const float*  __restrict__ query,
                                               const float*  __restrict__ ln_g,
                                               const float*  __restrict__ ln_b,
                                               const float*  __restrict__ Wp,
                                               const float*  __restrict__ bp,
                                               float* __restrict__ out) {
  __shared__ float sc[K_][T_];     // 24 KB
  __shared__ float qs[K_][H_];
  __shared__ float red[256];
  __shared__ float tln[K_][H_];
  const int tid = threadIdx.x;
  const int b   = blockIdx.x;
  const int len = lengths[b];
  for (int i = tid; i < K_ * H_; i += 256) qs[i / H_][i % H_] = query[i];
  __syncthreads();

  const float scale = 0.0625f;  // 256^-0.5
  const half_t* Hb = Hseq + (size_t)b * T_ * H_;

  for (int t = tid; t < len; t += 256) {
    const half_t* hr = Hb + (size_t)t * H_;
    float s0 = 0.f, s1 = 0.f, s2 = 0.f;
    for (int h = 0; h < H_; ++h) {
      const float v = (float)hr[h];
      s0 += qs[0][h] * v; s1 += qs[1][h] * v; s2 += qs[2][h] * v;
    }
    sc[0][t] = s0 * scale; sc[1][t] = s1 * scale; sc[2][t] = s2 * scale;
  }
  __syncthreads();

  float mx[K_], sm[K_];
  for (int k = 0; k < K_; ++k) {
    float lm = -3.0e38f;
    for (int t = tid; t < len; t += 256) lm = fmaxf(lm, sc[k][t]);
    mx[k] = block_max256(lm, red);
  }
  for (int k = 0; k < K_; ++k) {
    float ls = 0.f;
    for (int t = tid; t < len; t += 256) {
      const float e = expf(sc[k][t] - mx[k]);
      sc[k][t] = e;
      ls += e;
    }
    sm[k] = block_sum256(ls, red);   // barriers also publish sc writes
  }

  float a0 = 0.f, a1 = 0.f, a2 = 0.f;
  for (int t = 0; t < len; ++t) {
    const float hv = (float)Hb[(size_t)t * H_ + tid];
    a0 += sc[0][t] * hv; a1 += sc[1][t] * hv; a2 += sc[2][t] * hv;
  }
  float tk[K_] = { a0 / sm[0], a1 / sm[1], a2 / sm[2] };

  for (int k = 0; k < K_; ++k) {
    const float mean = block_sum256(tk[k], red) * (1.0f / H_);
    const float d    = tk[k] - mean;
    const float var  = block_sum256(d * d, red) * (1.0f / H_);
    tln[k][tid] = d * rsqrtf(var + 1e-5f) * ln_g[tid] + ln_b[tid];
  }
  __syncthreads();

  for (int k = 0; k < K_; ++k) {  // thread <-> output dim d
    float o = bp[tid];
    const float* wr = Wp + (size_t)tid * H_;
    for (int h = 0; h < H_; ++h) o += tln[k][h] * wr[h];
    out[((size_t)b * K_ + k) * D_ + tid] = o;
  }
}

// ------------------------------- launcher ----------------------------------

extern "C" void kernel_launch(void* const* d_in, const int* in_sizes, int n_in,
                              void* d_out, int out_size, void* d_ws, size_t ws_size,
                              hipStream_t stream) {
  const float* samples = (const float*)d_in[0];
  const int*   lengths = (const int*)d_in[1];
  const float* W1      = (const float*)d_in[2];
  const float* b1      = (const float*)d_in[3];
  const float* W2      = (const float*)d_in[4];
  const float* b2      = (const float*)d_in[5];
  const float* Wih     = (const float*)d_in[6];
  const float* bih     = (const float*)d_in[7];
  const float* Whh     = (const float*)d_in[8];
  const float* bhh     = (const float*)d_in[9];
  const float* query   = (const float*)d_in[10];
  const float* ln_g    = (const float*)d_in[11];
  const float* ln_b    = (const float*)d_in[12];
  const float* Wp      = (const float*)d_in[13];
  const float* bp      = (const float*)d_in[14];
  float* out = (float*)d_out;

  // Workspace layout (bytes):
  //   [0, 128K)       W2   f16
  //   [128K, 512K)    W_ih f16
  //   [512K, 896K)    W_hh f16
  //   [1M, 1M+403M)   xg   f16 [B*T, 768]
  //   [.., +134M)     x1 / Hseq f16 [B*T, 256]   (disjoint lifetimes)
  char* ws = (char*)d_ws;
  half_t* W2h  = (half_t*)(ws);
  half_t* Wihh = (half_t*)(ws + 131072);
  half_t* Whhh = (half_t*)(ws + 131072 + 393216);
  half_t* xg   = (half_t*)(ws + (1 << 20));
  half_t* x1   = (half_t*)(ws + (1 << 20) + (size_t)B_ * T_ * G3 * 2);
  half_t* Hseq = x1;

  k_cvt<<<(H_ * H_ + 255) / 256, 256, 0, stream>>>(W2, W2h, H_ * H_);
  k_cvt<<<(G3 * H_ + 255) / 256, 256, 0, stream>>>(Wih, Wihh, G3 * H_);
  k_cvt<<<(G3 * H_ + 255) / 256, 256, 0, stream>>>(Whh, Whhh, G3 * H_);

  k1_inproj1<<<B_ * T_, 256, 0, stream>>>(samples, W1, b1, x1);
  k2_x2_xg<<<B_ * T_ / 16, 256, 0, stream>>>(x1, W2h, b2, Wihh, bih, xg);
  k3_gru<<<B_ / 16, 256, 0, stream>>>(xg, Whhh, bhh, Hseq);
  k4_attn<<<B_, 256, 0, stream>>>(Hseq, lengths, query, ln_g, ln_b, Wp, bp, out);
}
